// TfmrAttention_78692390797635
// MI455X (gfx1250) — compile-verified, matrix operand order checked by reading, not run on hardware
//
#include <hip/hip_runtime.h>
#include <hip/hip_bf16.h>

typedef unsigned short u16;
typedef __attribute__((ext_vector_type(16))) __bf16 v16bf;
typedef __attribute__((ext_vector_type(8)))  float  v8f;

// ---------- bf16 bit helpers (RNE) ----------
__device__ __forceinline__ u16 f2bfbits(float f) {
  union { float f; unsigned u; } x; x.f = f;
  unsigned r = x.u + 0x7FFFu + ((x.u >> 16) & 1u);
  return (u16)(r >> 16);
}
__device__ __forceinline__ float bf2f(u16 s) {
  union { unsigned u; float f; } t; t.u = ((unsigned)s) << 16; return t.f;
}
__device__ __forceinline__ __bf16 bits2bf(u16 s) {
  union { u16 s; __bf16 b; } t; t.s = s; return t.b;
}
__device__ __forceinline__ u16 bf_bits_of(float v)  { return f2bfbits(v); }
__device__ __forceinline__ u16 bf_bits_of(u16 v)    { return v; }

// K index of fragment element e for lane-half h (16-bit A/B WMMA layout, ISA 7.12.2)
__device__ __forceinline__ int kOf(int e, int h) {
  return ((e & 8) << 1) + h * 8 + (e & 7);
}

__device__ __forceinline__ v8f wmma_bf16(v16bf a, v16bf b, v8f c) {
  return __builtin_amdgcn_wmma_f32_16x16x32_bf16(
      /*neg_a=*/false, a, /*neg_b=*/false, b,
      /*c_mod=*/(short)0, c, /*reuse_a=*/false, /*reuse_b=*/false);
}

__device__ __forceinline__ void store_out(float* p, float v) { *p = v; }
__device__ __forceinline__ void store_out(u16*   p, float v) { *p = f2bfbits(v); }

// ---------- fragment loads from padded LDS tiles ----------
__device__ __forceinline__ v16bf load_afrag_lds(const u16* base, int ld, int row0,
                                                int idx, int half) {
  v16bf a;
#pragma unroll
  for (int e = 0; e < 16; e += 2) {
    unsigned u = *(const unsigned*)(base + (row0 + idx) * ld + kOf(e, half));
    a[e]     = bits2bf((u16)(u & 0xffffu));
    a[e + 1] = bits2bf((u16)(u >> 16));
  }
  return a;
}
__device__ __forceinline__ v16bf load_bfrag_lds(const u16* base, int ld, int col0,
                                                int idx, int half) {
  v16bf b;
#pragma unroll
  for (int e = 0; e < 16; ++e)
    b[e] = bits2bf(base[kOf(e, half) * ld + col0 + idx]);
  return b;
}

// =====================================================================
// GEMM: Out[M,N] = bf16(A[M,K]) * bf16(W[K,N]) + bias, fp32 accumulate.
// WG = 256 threads = 8 waves; tile 128x64; wave tile 32x32 (2x2 WMMA).
// =====================================================================
#define KC 32
template <typename AT, typename OT>
__global__ __launch_bounds__(256)
void gemm_bias(const AT* __restrict__ A, const float* __restrict__ W,
               const float* __restrict__ bias, OT* __restrict__ Out,
               int M, int N, int K) {
  __shared__ u16 As[128][KC + 8];
  __shared__ u16 Bs[KC][64 + 8];

  const int tid  = threadIdx.x;
  const int lane = tid & 31;
  const int wave = tid >> 5;
  const int idx  = lane & 15;
  const int half = lane >> 4;
  const int wm   = wave >> 1;   // 0..3
  const int wn   = wave & 1;    // 0..1
  const int rowBase = blockIdx.y * 128;
  const int colBase = blockIdx.x * 64;

  v8f acc[2][2];
#pragma unroll
  for (int i = 0; i < 2; ++i)
#pragma unroll
    for (int j = 0; j < 2; ++j)
#pragma unroll
      for (int r = 0; r < 8; ++r) acc[i][j][r] = 0.0f;

  const int arow = tid >> 1, acol0 = (tid & 1) * 16;  // 128 rows x 32 k
  const int brow = tid >> 3, bcol0 = (tid & 7) * 8;   // 32 k x 64 cols

  for (int k0 = 0; k0 < K; k0 += KC) {
#pragma unroll
    for (int i = 0; i < 16; ++i)
      As[arow][acol0 + i] =
          bf_bits_of(A[(size_t)(rowBase + arow) * K + k0 + acol0 + i]);
#pragma unroll
    for (int i = 0; i < 8; ++i)
      Bs[brow][bcol0 + i] =
          f2bfbits(W[(size_t)(k0 + brow) * N + colBase + bcol0 + i]);
    __syncthreads();

    v16bf a0 = load_afrag_lds(&As[0][0], KC + 8, wm * 32,      idx, half);
    v16bf a1 = load_afrag_lds(&As[0][0], KC + 8, wm * 32 + 16, idx, half);
    v16bf b0 = load_bfrag_lds(&Bs[0][0], 64 + 8, wn * 32,      idx, half);
    v16bf b1 = load_bfrag_lds(&Bs[0][0], 64 + 8, wn * 32 + 16, idx, half);

    acc[0][0] = wmma_bf16(a0, b0, acc[0][0]);
    acc[0][1] = wmma_bf16(a0, b1, acc[0][1]);
    acc[1][0] = wmma_bf16(a1, b0, acc[1][0]);
    acc[1][1] = wmma_bf16(a1, b1, acc[1][1]);
    __syncthreads();
  }

#pragma unroll
  for (int i = 0; i < 2; ++i)
#pragma unroll
    for (int j = 0; j < 2; ++j) {
      const int col = colBase + wn * 32 + j * 16 + idx;
      const float bv = bias[col];
#pragma unroll
      for (int r = 0; r < 8; ++r) {
        const int row = rowBase + wm * 32 + i * 16 + r + 8 * half;
        store_out(&Out[(size_t)row * N + col], acc[i][j][r] + bv);
      }
    }
}

// =====================================================================
// Causal attention: one wave per (b, h, 16-row q tile).
// qkv: bf16 [B*S, 3E]; wts: fp32 [B,H,S,S]; ctx: bf16 [B*S, E]
// Pass A: streaming softmax stats (max, sumexp).
// Pass B: recompute scores, write normalized probs, O += P*V via WMMA.
// =====================================================================
__global__ __launch_bounds__(128)
void attn_kernel(const u16* __restrict__ qkv, float* __restrict__ wts,
                 u16* __restrict__ ctx) {
  constexpr int S = 2048, E = 1024, QS = 3 * E, D = 64;
  __shared__ u16 pscr[4][16][40];   // per-wave 16x32 P scratch (+pad)

  const int tid  = threadIdx.x;
  const int lane = tid & 31;
  const int wv   = tid >> 5;
  const int idx  = lane & 15;
  const int half = lane >> 4;

  const int gw    = blockIdx.x * 4 + wv;   // 0..4095
  const int qt    = gw & 127;
  const int h     = (gw >> 7) & 15;
  const int b     = gw >> 11;
  const int qbase = qt * 16;
  const int qlast = qbase + 15;

  const int qoff = h * D, koff = E + h * D, voff = 2 * E + h * D;

  // ---- Q fragments (pre-scaled by 1/sqrt(D) = 0.125, exact in bf16) ----
  v16bf qa[2];
#pragma unroll
  for (int c = 0; c < 2; ++c)
#pragma unroll
    for (int e = 0; e < 16; e += 2) {
      const int d = c * 32 + kOf(e, half);
      unsigned u = *(const unsigned*)(qkv +
          (size_t)(b * S + qbase + idx) * QS + qoff + d);
      qa[c][e]     = bits2bf(f2bfbits(bf2f((u16)(u & 0xffffu)) * 0.125f));
      qa[c][e + 1] = bits2bf(f2bfbits(bf2f((u16)(u >> 16))     * 0.125f));
    }

  // ---- Pass A: online stats over causal k tiles ----
  float m[8], l[8];
#pragma unroll
  for (int r = 0; r < 8; ++r) { m[r] = -3.0e38f; l[r] = 0.0f; }

  for (int kt = 0; kt <= qt; ++kt) {
    const int kb = kt * 16;
    v16bf kf[2];
#pragma unroll
    for (int c = 0; c < 2; ++c)
#pragma unroll
      for (int e = 0; e < 16; e += 2) {
        const int d = c * 32 + kOf(e, half);
        unsigned u = *(const unsigned*)(qkv +
            (size_t)(b * S + kb + idx) * QS + koff + d);
        kf[c][e]     = bits2bf((u16)(u & 0xffffu));
        kf[c][e + 1] = bits2bf((u16)(u >> 16));
      }
    v8f s;
#pragma unroll
    for (int r = 0; r < 8; ++r) s[r] = 0.0f;
    s = wmma_bf16(qa[0], kf[0], s);
    s = wmma_bf16(qa[1], kf[1], s);

    const int kcol = kb + idx;
#pragma unroll
    for (int r = 0; r < 8; ++r) {
      const int qrow = qbase + r + 8 * half;
      float v = (kcol <= qrow) ? s[r] : -3.0e38f;
      float tm = v;
      tm = fmaxf(tm, __shfl_xor(tm, 1, 32));
      tm = fmaxf(tm, __shfl_xor(tm, 2, 32));
      tm = fmaxf(tm, __shfl_xor(tm, 4, 32));
      tm = fmaxf(tm, __shfl_xor(tm, 8, 32));
      const float mn = fmaxf(m[r], tm);
      float p = __expf(v - mn);
      float ts = p;
      ts += __shfl_xor(ts, 1, 32);
      ts += __shfl_xor(ts, 2, 32);
      ts += __shfl_xor(ts, 4, 32);
      ts += __shfl_xor(ts, 8, 32);
      l[r] = l[r] * __expf(m[r] - mn) + ts;
      m[r] = mn;
    }
  }
  float rl[8];
#pragma unroll
  for (int r = 0; r < 8; ++r) rl[r] = 1.0f / l[r];

  // ---- Pass B: probabilities out + O = P*V ----
  float* wrow = wts + ((size_t)(b * 16 + h) * S + qbase) * S;
  v8f o[4];
#pragma unroll
  for (int nb = 0; nb < 4; ++nb)
#pragma unroll
    for (int r = 0; r < 8; ++r) o[nb][r] = 0.0f;

  u16 (*ps)[40] = pscr[wv];

  for (int kp = 0; kp < 64; ++kp) {
    const int kb = kp * 32;
    if (kb > qlast) {  // fully masked pair: zero-fill weights only
#pragma unroll
      for (int t = 0; t < 2; ++t)
#pragma unroll
        for (int r = 0; r < 8; ++r)
          wrow[(size_t)(r + 8 * half) * S + kb + t * 16 + idx] = 0.0f;
      continue;
    }
#pragma unroll
    for (int t = 0; t < 2; ++t) {
      const int kb2 = kb + t * 16;
      if (kb2 > qlast) {  // masked half of the pair
#pragma unroll
        for (int r = 0; r < 8; ++r) {
          wrow[(size_t)(r + 8 * half) * S + kb2 + idx] = 0.0f;
          ps[r + 8 * half][t * 16 + idx] = 0;
        }
        continue;
      }
      v16bf kf[2];
#pragma unroll
      for (int c = 0; c < 2; ++c)
#pragma unroll
        for (int e = 0; e < 16; e += 2) {
          const int d = c * 32 + kOf(e, half);
          unsigned u = *(const unsigned*)(qkv +
              (size_t)(b * S + kb2 + idx) * QS + koff + d);
          kf[c][e]     = bits2bf((u16)(u & 0xffffu));
          kf[c][e + 1] = bits2bf((u16)(u >> 16));
        }
      v8f s;
#pragma unroll
      for (int r = 0; r < 8; ++r) s[r] = 0.0f;
      s = wmma_bf16(qa[0], kf[0], s);
      s = wmma_bf16(qa[1], kf[1], s);

      const int kcol = kb2 + idx;
#pragma unroll
      for (int r = 0; r < 8; ++r) {
        const int qrow = qbase + r + 8 * half;
        const float v = (kcol <= qrow) ? s[r] : -3.0e38f;
        const float p = __expf(v - m[r]) * rl[r];
        wrow[(size_t)(r + 8 * half) * S + kcol] = p;
        ps[r + 8 * half][t * 16 + idx] = f2bfbits(p);
      }
    }
    asm volatile("s_wait_dscnt 0" ::: "memory");  // DS in-order; fence compiler too

    v16bf pa;
#pragma unroll
    for (int e = 0; e < 16; e += 2) {
      unsigned u = *(const unsigned*)(&ps[idx][kOf(e, half)]);
      pa[e]     = bits2bf((u16)(u & 0xffffu));
      pa[e + 1] = bits2bf((u16)(u >> 16));
    }
#pragma unroll
    for (int nb = 0; nb < 4; ++nb) {
      v16bf vb;
#pragma unroll
      for (int e = 0; e < 16; ++e) {
        const int key = kb + kOf(e, half);
        vb[e] = bits2bf(qkv[(size_t)(b * S + key) * QS + voff + nb * 16 + idx]);
      }
      o[nb] = wmma_bf16(pa, vb, o[nb]);
    }
    asm volatile("s_wait_dscnt 0" ::: "memory");
  }

#pragma unroll
  for (int nb = 0; nb < 4; ++nb)
#pragma unroll
    for (int r = 0; r < 8; ++r)
      ctx[(size_t)(b * S + qbase + r + 8 * half) * E + h * D + nb * 16 + idx] =
          f2bfbits(o[nb][r]);
}

// =====================================================================
// Host-side launcher
// =====================================================================
extern "C" void kernel_launch(void* const* d_in, const int* in_sizes, int n_in,
                              void* d_out, int out_size, void* d_ws, size_t ws_size,
                              hipStream_t stream) {
  (void)in_sizes; (void)n_in; (void)out_size; (void)ws_size;
  const float* hs        = (const float*)d_in[0];  // [2,2048,1024]
  const float* c_attn_w  = (const float*)d_in[1];  // [1024,3072]
  const float* c_attn_b  = (const float*)d_in[2];  // [3072]
  const float* c_proj_w  = (const float*)d_in[3];  // [1024,1024]
  const float* c_proj_b  = (const float*)d_in[4];  // [1024]

  float* attn_out = (float*)d_out;                          // 4096*1024 f32
  float* attn_w   = (float*)d_out + (size_t)4096 * 1024;    // 2*16*2048*2048 f32

  u16* qkv = (u16*)d_ws;                                            // 24 MB bf16
  u16* ctx = (u16*)((char*)d_ws + (size_t)4096 * 3072 * sizeof(u16)); // 8 MB bf16

  // 1) QKV = bf16(X @ Wqkv + b)
  gemm_bias<float, u16><<<dim3(3072 / 64, 4096 / 128), 256, 0, stream>>>(
      hs, c_attn_w, c_attn_b, qkv, 4096, 3072, 1024);

  // 2) causal attention: probs -> d_out weights region, ctx -> ws
  attn_kernel<<<dim3(1024), 128, 0, stream>>>(qkv, attn_w, ctx);

  // 3) attn_out = ctx @ Wproj + b
  gemm_bias<u16, float><<<dim3(1024 / 64, 4096 / 128), 256, 0, stream>>>(
      ctx, c_proj_w, c_proj_b, attn_out, 4096, 1024, 1024);
}
